// TransformerBlock_69947837382852
// MI455X (gfx1250) — matrix operand order repaired and checked by reading, not compiled, not run on hardware
//
#include <hip/hip_runtime.h>
#include <hip/hip_bf16.h>

// ---------------------------------------------------------------------------
// Transformer block (pre-norm attn + FFN) for gfx1250: bf16 WMMA + TDM.
// ---------------------------------------------------------------------------

typedef __bf16 bf16_t;
typedef __attribute__((ext_vector_type(16))) __bf16 v16bf;
typedef __attribute__((ext_vector_type(8)))  __bf16 v8bf;
typedef __attribute__((ext_vector_type(8)))  float  v8f;

#define DMODEL 1024
#define NTOK   4096   // B*N = 2*2048
#define SEQ    2048
#define NHEAD  16
#define HDIM   64
#define FFDIM  4096

#define BK      128   // K-chunk staged in LDS per TDM transfer
#define LDB_PAD 136   // LDS row stride in halves: 128 + 8 (272B -> conflict-free)

#if __has_builtin(__builtin_amdgcn_tensor_load_to_lds)
#define HAVE_TDM 1
#else
#define HAVE_TDM 0
#endif

// float -> bf16, round-to-nearest-even
__device__ __forceinline__ bf16_t f2bf(float f) {
  union { float f; unsigned u; } in; in.f = f;
  unsigned u = in.u;
  u += 0x7fffu + ((u >> 16) & 1u);
  union { unsigned short s; bf16_t b; } out;
  out.s = (unsigned short)(u >> 16);
  return out.b;
}

__device__ __forceinline__ v8f wmma_bf16(v16bf a, v16bf b, v8f c) {
  // D = A(16x32) * B(32x16) + C, fp32 accumulate
  return __builtin_amdgcn_wmma_f32_16x16x32_bf16(false, a, false, b, (short)0, c,
                                                 false, false);
}

// A fragment: 16(M) x 32(K) from row-major [*, lda] source.
// Lane L: row M = m + (L&15); halves 0..7 = K(k0+kb..+7), 8..15 = K(k0+16+kb..+23),
// kb = (L<16)?0:8   (ISA 16-bit A layout)
__device__ __forceinline__ v16bf load_frag_a(const bf16_t* A, int lda, int m,
                                             int k0, int lane) {
  int mm = m + (lane & 15);
  int kb = (lane & 16) ? 8 : 0;
  const bf16_t* p = A + (size_t)mm * lda + k0 + kb;
  v8bf lo = *(const v8bf*)p;
  v8bf hi = *(const v8bf*)(p + 16);
  return __builtin_shufflevector(lo, hi, 0,1,2,3,4,5,6,7,8,9,10,11,12,13,14,15);
}

// B fragment: 32(K) x 16(N); column n of B = row n of row-major W[N,K].
// Lane L: col N = n + (L&15); halves 0..7 = K(k0+kb..+7), 8..15 = K(k0+16+kb..+23), kb = (L<16)?0:8.
__device__ __forceinline__ v16bf load_frag_b(const bf16_t* W, int ldw, int n,
                                             int k0, int lane) {
  int nn = n + (lane & 15);
  int kb = (lane & 16) ? 8 : 0;
  const bf16_t* p = W + (size_t)nn * ldw + k0 + kb;
  v8bf lo = *(const v8bf*)p;
  v8bf hi = *(const v8bf*)(p + 16);
  return __builtin_shufflevector(lo, hi, 0,1,2,3,4,5,6,7,8,9,10,11,12,13,14,15);
}

#if HAVE_TDM
// Issue one TDM transfer: 64 rows x BK halves of W (row-major [Nrows, K]) into
// LDS at lds_addr, padded 4 DWORDs per 256B row (LDS row stride = LDB_PAD halves).
// Descriptor per CDNA5 ISA. This toolchain's
// builtin takes 6 args: (g0 v4u, g1 v8i, g2 v4i, g3 v4i, v8i, i32 cpol).
__device__ __forceinline__ void tdm_load_b_tile(const bf16_t* W, int K, int Nrows,
                                                int Nbase, int kb, unsigned lds_addr) {
  typedef unsigned v4u __attribute__((ext_vector_type(4)));
  typedef int      v4i __attribute__((ext_vector_type(4)));
  typedef int      v8i __attribute__((ext_vector_type(8)));
  unsigned long long ga =
      (unsigned long long)(size_t)(W + (size_t)Nbase * K + (size_t)kb);
  v4u g0;
  g0[0] = 1u;                                                // count=1 (valid), user mode
  g0[1] = lds_addr;                                          // lds_addr[31:0]
  g0[2] = (unsigned)(ga & 0xffffffffu);                      // global_addr[31:0]
  g0[3] = (unsigned)((ga >> 32) & 0x01ffffffu) | 0x80000000u;// addr[56:32] | type=2
  v8i g1;
  g1[0] = (1 << 16)            // data_size = 1 -> 2 bytes
        | (1 << 20)            // pad_enable
        | (5 << 22)            // pad_interval = 5 -> every 64 DWORDs (256B row)
        | (3 << 25);           // pad_amount = 3 -> 4 DWORDs (row stride 272B)
  g1[1] = (K & 0xffff) << 16;                                // tensor_dim0[15:0]
  g1[2] = ((K >> 16) & 0xffff) | ((Nrows & 0xffff) << 16);   // dim0[31:16] | dim1[15:0]
  g1[3] = ((Nrows >> 16) & 0xffff) | (BK << 16);             // dim1[31:16] | tile_dim0=BK
  g1[4] = 64;                                                // tile_dim1=64, tile_dim2=0
  g1[5] = K;                                                 // tensor_dim0_stride[31:0]
  g1[6] = 0;                                                 // stride[47:32], dim1_stride lo
  g1[7] = 0;
  v4i gz  = {0, 0, 0, 0};                                    // dims 2..4 unused
  v8i gz8 = {0, 0, 0, 0, 0, 0, 0, 0};
  __builtin_amdgcn_tensor_load_to_lds(g0, g1, gz, gz, gz8, 0);
}
#endif

// ---------------------------------------------------------------------------
__global__ void cvt_f32_bf16_kernel(const float* __restrict__ in,
                                    bf16_t* __restrict__ out, int n) {
  int i = blockIdx.x * blockDim.x + threadIdx.x;
  if (i < n) out[i] = f2bf(in[i]);
}

// ---------------------------------------------------------------------------
// LayerNorm over D=1024, one block (256 threads) per token, bf16 output.
__global__ __launch_bounds__(256) void ln_bf16_kernel(
    const float* __restrict__ x, const float* __restrict__ g,
    const float* __restrict__ b, bf16_t* __restrict__ out) {
  int t = blockIdx.x;
  int tid = threadIdx.x;
  const float4 v = ((const float4*)(x + (size_t)t * DMODEL))[tid];
  float s  = v.x + v.y + v.z + v.w;
  float s2 = v.x*v.x + v.y*v.y + v.z*v.z + v.w*v.w;
#pragma unroll
  for (int off = 16; off > 0; off >>= 1) {
    s  += __shfl_xor(s,  off, 32);
    s2 += __shfl_xor(s2, off, 32);
  }
  __shared__ float ws1[8], ws2[8];
  int wid = tid >> 5, lane = tid & 31;
  if (lane == 0) { ws1[wid] = s; ws2[wid] = s2; }
  __syncthreads();
  float ts = 0.f, ts2 = 0.f;
#pragma unroll
  for (int i = 0; i < 8; i++) { ts += ws1[i]; ts2 += ws2[i]; }
  float mean = ts * (1.0f / DMODEL);
  float var  = ts2 * (1.0f / DMODEL) - mean * mean;
  float rstd = rsqrtf(var + 1e-5f);
  const float4 gv = ((const float4*)g)[tid];
  const float4 bv = ((const float4*)b)[tid];
  bf16_t* op = out + (size_t)t * DMODEL + tid * 4;
  op[0] = f2bf((v.x - mean) * rstd * gv.x + bv.x);
  op[1] = f2bf((v.y - mean) * rstd * gv.y + bv.y);
  op[2] = f2bf((v.z - mean) * rstd * gv.z + bv.z);
  op[3] = f2bf((v.w - mean) * rstd * gv.w + bv.w);
}

// ---------------------------------------------------------------------------
// GEMM: Y[t,n] = sum_k A[t,k]*W[n,k] + bias[n] (+resid) (opt. GELU)
// Block = 256 threads = 8 waves; wave: 32(M) x 64(N); block: 256 x 64.
// B tile (64 N-rows x BK K) staged in LDS by the Tensor Data Mover, double
// buffered; A fragments stream straight from global (per-wave-unique rows).
// mode 0: fp32 out, 1: bf16 out, 2: scatter to q(scaled)/k/vT.
__global__ __launch_bounds__(256) void gemm_bf16_kernel(
    const bf16_t* __restrict__ A, const bf16_t* __restrict__ W,
    const float* __restrict__ bias, const float* __restrict__ resid,
    int K, int Ncols, int mode, int act,
    float* __restrict__ outf, bf16_t* __restrict__ outb,
    bf16_t* __restrict__ qout, bf16_t* __restrict__ kout,
    bf16_t* __restrict__ vtout) {
  int tid = threadIdx.x;
  int wid = tid >> 5, lane = tid & 31;
  int Mbase = blockIdx.x * 256 + wid * 32;
  int Nbase = blockIdx.y * 64;

  __shared__ bf16_t shB[2][64 * LDB_PAD];   // double-buffered padded B tile

  v8f acc[2][4];
#pragma unroll
  for (int mt = 0; mt < 2; mt++)
#pragma unroll
    for (int nt = 0; nt < 4; nt++)
      acc[mt][nt] = (v8f){0.f,0.f,0.f,0.f,0.f,0.f,0.f,0.f};

#if HAVE_TDM
  if (wid == 0)   // one wave drives the TDM (EXEC ignored by tensor ops)
    tdm_load_b_tile(W, K, Ncols, Nbase, 0, (unsigned)(size_t)&shB[0][0]);
#endif

  for (int kb = 0; kb < K; kb += BK) {
    int bi = (kb / BK) & 1;
#if HAVE_TDM
    if (wid == 0) {
      if (kb + BK < K) {
        tdm_load_b_tile(W, K, Ncols, Nbase, kb + BK,
                        (unsigned)(size_t)&shB[1 - bi][0]);
        __builtin_amdgcn_s_wait_tensorcnt(1);   // current tile landed
      } else {
        __builtin_amdgcn_s_wait_tensorcnt(0);
      }
    }
    __syncthreads();                            // publish LDS tile to all waves
#else
    // cooperative fallback: same padded layout, vectorized copy
    for (int i = tid; i < 64 * (BK / 8); i += 256) {
      int r = i / (BK / 8), c = (i % (BK / 8)) * 8;
      *(v8bf*)&shB[bi][r * LDB_PAD + c] =
          *(const v8bf*)(W + (size_t)(Nbase + r) * K + kb + c);
    }
    __syncthreads();
#endif
    const bf16_t* Wl = &shB[bi][0];
#pragma unroll
    for (int k0 = 0; k0 < BK; k0 += 32) {
      v16bf a0 = load_frag_a(A, K, Mbase,      kb + k0, lane);
      v16bf a1 = load_frag_a(A, K, Mbase + 16, kb + k0, lane);
#pragma unroll
      for (int nt = 0; nt < 4; nt++) {
        v16bf bfr = load_frag_b(Wl, LDB_PAD, nt * 16, k0, lane);
        acc[0][nt] = wmma_bf16(a0, bfr, acc[0][nt]);
        acc[1][nt] = wmma_bf16(a1, bfr, acc[1][nt]);
      }
    }
    __syncthreads();   // all waves done with buf[bi] before TDM overwrites it
  }

  int rowoff = (lane & 16) ? 8 : 0;
  int nlo = lane & 15;
#pragma unroll
  for (int mt = 0; mt < 2; mt++) {
#pragma unroll
    for (int nt = 0; nt < 4; nt++) {
      int gn = Nbase + nt * 16 + nlo;
      float bz = bias ? bias[gn] : 0.0f;
#pragma unroll
      for (int j = 0; j < 8; j++) {
        int gm = Mbase + mt * 16 + rowoff + j;
        float v = acc[mt][nt][j] + bz;
        if (act == 1) v = 0.5f * v * (1.0f + erff(v * 0.70710678118f));
        if (resid) v += resid[(size_t)gm * Ncols + gn];
        if (mode == 0) {
          outf[(size_t)gm * Ncols + gn] = v;
        } else if (mode == 1) {
          outb[(size_t)gm * Ncols + gn] = f2bf(v);
        } else {
          int s = gn >> 10, r = gn & 1023, hh = r >> 6, dd = r & 63;
          int bb = gm >> 11, tn = gm & (SEQ - 1);
          size_t bhn = ((size_t)(bb * NHEAD + hh)) * SEQ + tn;
          if (s == 0)      qout[bhn * HDIM + dd] = f2bf(v * 0.125f); // * HD^-0.5
          else if (s == 1) kout[bhn * HDIM + dd] = f2bf(v);
          else vtout[(((size_t)(bb * NHEAD + hh)) * HDIM + dd) * SEQ + tn] = f2bf(v);
        }
      }
    }
  }
}

// ---------------------------------------------------------------------------
// Flash-style attention. Grid.x = B*H*(SEQ/128); block = 256 = 8 waves;
// each wave owns 16 queries, streams keys/values in 32-key blocks.
// q/k: [B,H,N,64] bf16 (q pre-scaled); vT: [B,H,64,N]; o: [B*N, 1024] bf16.
__global__ __launch_bounds__(256) void attn_kernel(
    const bf16_t* __restrict__ q, const bf16_t* __restrict__ k,
    const bf16_t* __restrict__ vt, bf16_t* __restrict__ o) {
  int wid = threadIdx.x >> 5, lane = threadIdx.x & 31;
  int bh = blockIdx.x >> 4;       // 0..31 = b*16+h
  int qblk = blockIdx.x & 15;
  int qbase = qblk * 128 + wid * 16;
  int bb = bh >> 4, hh = bh & 15;
  const bf16_t* qp = q  + (size_t)bh * SEQ * HDIM;
  const bf16_t* kp = k  + (size_t)bh * SEQ * HDIM;
  const bf16_t* vp = vt + (size_t)bh * HDIM * SEQ;

  // P staging: per-wave 16x32 bf16 tile, row stride 40 halves (16B aligned)
  __shared__ bf16_t psh[8][16 * 40];
  bf16_t* pw = psh[wid];

  float mrun[8], lrun[8];
  v8f accO[4];
#pragma unroll
  for (int j = 0; j < 8; j++) { mrun[j] = -1e30f; lrun[j] = 0.0f; }
#pragma unroll
  for (int dt = 0; dt < 4; dt++)
    accO[dt] = (v8f){0.f,0.f,0.f,0.f,0.f,0.f,0.f,0.f};

  int half8 = (lane & 16) ? 8 : 0;
  int nlo = lane & 15;

  for (int kb = 0; kb < SEQ; kb += 32) {
    // --- S = q . k^T over HD=64 (scale folded into q) ---
    v8f s0 = (v8f){0.f,0.f,0.f,0.f,0.f,0.f,0.f,0.f};
    v8f s1 = (v8f){0.f,0.f,0.f,0.f,0.f,0.f,0.f,0.f};
#pragma unroll
    for (int kk = 0; kk < HDIM; kk += 32) {
      v16bf aq = load_frag_a(qp, HDIM, qbase, kk, lane);
      v16bf b0 = load_frag_b(kp, HDIM, kb,      kk, lane);
      v16bf b1 = load_frag_b(kp, HDIM, kb + 16, kk, lane);
      s0 = wmma_bf16(aq, b0, s0);
      s1 = wmma_bf16(aq, b1, s1);
    }
    // --- online softmax (rows live in (half, VGPR j); reduce across 16 lanes) ---
#pragma unroll
    for (int j = 0; j < 8; j++) {
      float mx = fmaxf(s0[j], s1[j]);
#pragma unroll
      for (int off = 8; off >= 1; off >>= 1)
        mx = fmaxf(mx, __shfl_xor(mx, off, 32));
      float mtot = fmaxf(mrun[j], mx);
      float fs = __expf(mrun[j] - mtot);
      mrun[j] = mtot;
      float p0 = __expf(s0[j] - mtot);
      float p1 = __expf(s1[j] - mtot);
      float rs = p0 + p1;
#pragma unroll
      for (int off = 8; off >= 1; off >>= 1)
        rs += __shfl_xor(rs, off, 32);
      lrun[j] = lrun[j] * fs + rs;
      accO[0][j] *= fs; accO[1][j] *= fs; accO[2][j] *= fs; accO[3][j] *= fs;
      int row = j + half8;
      pw[row * 40 + nlo]      = f2bf(p0);
      pw[row * 40 + 16 + nlo] = f2bf(p1);
    }
    // same-wave LDS RAW: make the P tile visible before transposed reload
    asm volatile("s_wait_dscnt 0x0" ::: "memory");
    // --- O += P(16x32) . V(32x64) ---
    v16bf ap = load_frag_a(pw, 40, 0, 0, lane);
#pragma unroll
    for (int dt = 0; dt < 4; dt++) {
      v16bf bv = load_frag_b(vp, SEQ, dt * 16, kb, lane);
      accO[dt] = wmma_bf16(ap, bv, accO[dt]);
    }
  }

  // epilogue: o[b, n, h*64+d] bf16
#pragma unroll
  for (int dt = 0; dt < 4; dt++) {
#pragma unroll
    for (int j = 0; j < 8; j++) {
      int row = half8 + j;
      int t = bb * SEQ + qbase + row;
      int c = hh * HDIM + dt * 16 + nlo;
      o[(size_t)t * DMODEL + c] = f2bf(accO[dt][j] / lrun[j]);
    }
  }
}

// ---------------------------------------------------------------------------
extern "C" void kernel_launch(void* const* d_in, const int* in_sizes, int n_in,
                              void* d_out, int out_size, void* d_ws, size_t ws_size,
                              hipStream_t stream) {
  const float* x      = (const float*)d_in[0];
  const float* ln1_g  = (const float*)d_in[1];
  const float* ln1_b  = (const float*)d_in[2];
  const float* w_qkv  = (const float*)d_in[3];
  const float* b_qkv  = (const float*)d_in[4];
  const float* w_proj = (const float*)d_in[5];
  const float* b_proj = (const float*)d_in[6];
  const float* ln2_g  = (const float*)d_in[7];
  const float* ln2_b  = (const float*)d_in[8];
  const float* w_fc1  = (const float*)d_in[9];
  const float* b_fc1  = (const float*)d_in[10];
  const float* w_fc2  = (const float*)d_in[11];
  const float* b_fc2  = (const float*)d_in[12];
  float* out = (float*)d_out;

  char* ws = (char*)d_ws;
  size_t off = 0;
  auto alloc = [&](size_t bytes) -> char* {
    char* p = ws + off;
    off += (bytes + 255) & ~(size_t)255;
    return p;
  };
  const int D = DMODEL, FF = FFDIM, T = NTOK;
  bf16_t* wqkv_b  = (bf16_t*)alloc((size_t)3 * D * D * 2);
  bf16_t* wproj_b = (bf16_t*)alloc((size_t)D * D * 2);
  bf16_t* wfc1_b  = (bf16_t*)alloc((size_t)FF * D * 2);
  bf16_t* wfc2_b  = (bf16_t*)alloc((size_t)D * FF * 2);
  bf16_t* h_b     = (bf16_t*)alloc((size_t)T * D * 2);   // LN1 then LN2 output
  bf16_t* q_b     = (bf16_t*)alloc((size_t)T * D * 2);
  bf16_t* k_b     = (bf16_t*)alloc((size_t)T * D * 2);
  bf16_t* vt_b    = (bf16_t*)alloc((size_t)T * D * 2);
  bf16_t* o_b     = (bf16_t*)alloc((size_t)T * D * 2);
  float*  x1_f    = (float*) alloc((size_t)T * D * 4);   // post-attn residual
  bf16_t* g_b     = (bf16_t*)alloc((size_t)T * FF * 2);  // FFN intermediate

  // --- weight conversion fp32 -> bf16 ---
  cvt_f32_bf16_kernel<<<(3 * D * D + 255) / 256, 256, 0, stream>>>(w_qkv,  wqkv_b,  3 * D * D);
  cvt_f32_bf16_kernel<<<(D * D + 255) / 256,     256, 0, stream>>>(w_proj, wproj_b, D * D);
  cvt_f32_bf16_kernel<<<(FF * D + 255) / 256,    256, 0, stream>>>(w_fc1,  wfc1_b,  FF * D);
  cvt_f32_bf16_kernel<<<(D * FF + 255) / 256,    256, 0, stream>>>(w_fc2,  wfc2_b,  D * FF);

  // --- h = LN1(x) ---
  ln_bf16_kernel<<<T, 256, 0, stream>>>(x, ln1_g, ln1_b, h_b);

  // --- qkv = h @ w_qkv^T + b_qkv, scattered to q(scaled)/k/vT ---
  {
    dim3 grid(T / 256, (3 * D) / 64);
    gemm_bf16_kernel<<<grid, 256, 0, stream>>>(h_b, wqkv_b, b_qkv, nullptr,
                                               D, 3 * D, /*mode*/2, /*act*/0,
                                               nullptr, nullptr, q_b, k_b, vt_b);
  }

  // --- attention -> o (bf16 [T, D]) ---
  attn_kernel<<<32 * (SEQ / 128), 256, 0, stream>>>(q_b, k_b, vt_b, o_b);

  // --- x1 = x + o @ w_proj^T + b_proj (fp32) ---
  {
    dim3 grid(T / 256, D / 64);
    gemm_bf16_kernel<<<grid, 256, 0, stream>>>(o_b, wproj_b, b_proj, x,
                                               D, D, /*mode*/0, /*act*/0,
                                               x1_f, nullptr, nullptr, nullptr, nullptr);
  }

  // --- h2 = LN2(x1) ---
  ln_bf16_kernel<<<T, 256, 0, stream>>>(x1_f, ln2_g, ln2_b, h_b);

  // --- g = gelu(h2 @ w_fc1^T + b_fc1) (bf16) ---
  {
    dim3 grid(T / 256, FF / 64);
    gemm_bf16_kernel<<<grid, 256, 0, stream>>>(h_b, wfc1_b, b_fc1, nullptr,
                                               D, FF, /*mode*/1, /*act*/1,
                                               nullptr, g_b, nullptr, nullptr, nullptr);
  }

  // --- out = x1 + g @ w_fc2^T + b_fc2 (fp32) ---
  {
    dim3 grid(T / 256, D / 64);
    gemm_bf16_kernel<<<grid, 256, 0, stream>>>(g_b, wfc2_b, b_fc2, x1_f,
                                               FF, D, /*mode*/0, /*act*/0,
                                               out, nullptr, nullptr, nullptr, nullptr);
  }
}
